// IdpGANGenerator_75093208203304
// MI455X (gfx1250) — matrix operand order repaired and checked by reading, not compiled, not run on hardware
//
#include <hip/hip_runtime.h>
#include <hip/hip_bf16.h>

// ---------------- model dims ----------------
#define BB 8
#define NZ 16
#define LL 512
#define EE 256
#define DMOD 512
#define NHH 16
#define HDD 32
#define FFF 1024
#define NLL 8
#define E1DD 32
#define PDIMM 64
#define PMAXX 24
#define ROWS (BB*LL)            // 4096
#define NBIN (2*PMAXX+1)        // 49

typedef __attribute__((ext_vector_type(16))) _Float16 v16h;
typedef __attribute__((ext_vector_type(8)))  _Float16 v8h;
typedef __attribute__((ext_vector_type(8)))  float    v8f;

// ---------------- WMMA helpers ----------------
__device__ __forceinline__ v8f wmma_f16(v16h a, v16h b, v8f c) {
  return __builtin_amdgcn_wmma_f32_16x16x32_f16(false, a, false, b, (short)0, c,
                                                false, false);
}

// A-operand fragment: lane = g*16+m holds row m, K = 16g + j
// -> 16 contiguous floats at [16g, 16g+16)
__device__ __forceinline__ v16h load_a16(const float* __restrict__ p, int g) {
  const float4* a = (const float4*)(p + 16 * g);
  float4 x0 = a[0], x1 = a[1];
  const float4* b = a + 2;
  float4 x2 = b[0], x3 = b[1];
  v16h r;
  r[0]=(_Float16)x0.x; r[1]=(_Float16)x0.y; r[2]=(_Float16)x0.z; r[3]=(_Float16)x0.w;
  r[4]=(_Float16)x1.x; r[5]=(_Float16)x1.y; r[6]=(_Float16)x1.z; r[7]=(_Float16)x1.w;
  r[8]=(_Float16)x2.x; r[9]=(_Float16)x2.y; r[10]=(_Float16)x2.z; r[11]=(_Float16)x2.w;
  r[12]=(_Float16)x3.x; r[13]=(_Float16)x3.y; r[14]=(_Float16)x3.z; r[15]=(_Float16)x3.w;
  return r;
}

// B-operand fragment from f32 row: lane = g*16+n holds col n, K = 16g + j
// -> 16 contiguous floats at [16g, 16g+16)
__device__ __forceinline__ v16h load_b16_f32(const float* __restrict__ p, int g,
                                             float sc) {
  const float4* a = (const float4*)(p + 16 * g);
  float4 x0 = a[0], x1 = a[1], x2 = a[2], x3 = a[3];
  v16h r;
  r[0]=(_Float16)(x0.x*sc); r[1]=(_Float16)(x0.y*sc); r[2]=(_Float16)(x0.z*sc); r[3]=(_Float16)(x0.w*sc);
  r[4]=(_Float16)(x1.x*sc); r[5]=(_Float16)(x1.y*sc); r[6]=(_Float16)(x1.z*sc); r[7]=(_Float16)(x1.w*sc);
  r[8]=(_Float16)(x2.x*sc); r[9]=(_Float16)(x2.y*sc); r[10]=(_Float16)(x2.z*sc); r[11]=(_Float16)(x2.w*sc);
  r[12]=(_Float16)(x3.x*sc); r[13]=(_Float16)(x3.y*sc); r[14]=(_Float16)(x3.z*sc); r[15]=(_Float16)(x3.w*sc);
  return r;
}

// B-operand fragment from pre-transposed f16 weights (row n contiguous over K)
__device__ __forceinline__ v16h load_bt(const _Float16* __restrict__ bp) {
  v8h lo = *(const v8h*)bp;
  v8h hi = *(const v8h*)(bp + 8);
  v16h r;
#pragma unroll
  for (int i = 0; i < 8; ++i) { r[i] = lo[i]; r[8 + i] = hi[i]; }
  return r;
}

// ---------------- generic WMMA GEMM: C(MxN) = act(A(MxK) @ BtT + bias + resid)
// one wave computes a 32x64 tile (2 A fragments x 4 shared B fragments =
// 8 WMMAs per K-step); Bt is (N,K) f16 row-major.
__device__ __forceinline__ void store8(float* __restrict__ C, int ldc,
                                       const float* __restrict__ resid,
                                       int relu, int rowbase, int col,
                                       float bb, v8f acc) {
#pragma unroll
  for (int r = 0; r < 8; ++r) {
    int row = rowbase + r;
    float val = acc[r] + bb;
    if (resid) val += resid[(size_t)row * ldc + col];
    if (relu) val = fmaxf(val, 0.0f);
    C[(size_t)row * ldc + col] = val;
  }
}

__global__ void __launch_bounds__(32)
gemm_wmma_kernel(const float* __restrict__ A, int lda,
                 const _Float16* __restrict__ Bt, int ldb,
                 float* __restrict__ C, int ldc,
                 const float* __restrict__ bias,
                 const float* __restrict__ resid, int K, int relu) {
  int lane = threadIdx.x;
  int g = lane >> 4, nn = lane & 15;
  int row0 = blockIdx.y << 5;   // 32 rows
  int col0 = blockIdx.x << 6;   // 64 cols
  const float* arow0 = A + (size_t)(row0 + nn) * lda;
  const float* arow1 = arow0 + (size_t)16 * lda;
  const _Float16* bp0 = Bt + (size_t)(col0 + nn) * ldb;
  const _Float16* bp1 = bp0 + (size_t)16 * ldb;
  const _Float16* bp2 = bp0 + (size_t)32 * ldb;
  const _Float16* bp3 = bp0 + (size_t)48 * ldb;
  v8f a0 = {}, a1 = {}, a2 = {}, a3 = {};   // rows row0..row0+15
  v8f c0 = {}, c1 = {}, c2 = {}, c3 = {};   // rows row0+16..row0+31
  for (int kk = 0; kk < K; kk += 32) {
    v16h avA = load_a16(arow0 + kk, g);
    v16h avB = load_a16(arow1 + kk, g);
    int ko = kk + 16 * g;
    v16h b0 = load_bt(bp0 + ko);
    v16h b1 = load_bt(bp1 + ko);
    v16h b2f = load_bt(bp2 + ko);
    v16h b3 = load_bt(bp3 + ko);
    a0 = wmma_f16(avA, b0, a0);
    c0 = wmma_f16(avB, b0, c0);
    a1 = wmma_f16(avA, b1, a1);
    c1 = wmma_f16(avB, b1, c1);
    a2 = wmma_f16(avA, b2f, a2);
    c2 = wmma_f16(avB, b2f, c2);
    a3 = wmma_f16(avA, b3, a3);
    c3 = wmma_f16(avB, b3, c3);
  }
  int rb0 = row0 + 8 * g;
  int rb1 = row0 + 16 + 8 * g;
#pragma unroll
  for (int t = 0; t < 4; ++t) {
    int col = col0 + 16 * t + nn;
    float bb = bias ? bias[col] : 0.0f;
    v8f lo = (t == 0) ? a0 : (t == 1) ? a1 : (t == 2) ? a2 : a3;
    v8f hi = (t == 0) ? c0 : (t == 1) ? c1 : (t == 2) ? c2 : c3;
    store8(C, ldc, resid, relu, rb0, col, bb, lo);
    store8(C, ldc, resid, relu, rb1, col, bb, hi);
  }
}

// ---------------- fused flash attention (one wave = 16 queries, 1 head) -----
// q,k,v: (B,L,NH,HD) f32.  lut: 49 bins x 16 heads for this layer.
__global__ void __launch_bounds__(32)
attn_kernel(const float* __restrict__ q, const float* __restrict__ k,
            const float* __restrict__ v, const float* __restrict__ lut,
            float* __restrict__ o) {
  __shared__ float sl[NBIN];
  int lane = threadIdx.x;
  int qt = blockIdx.x, hh = blockIdx.y, b = blockIdx.z;
  for (int i = lane; i < NBIN; i += 32) sl[i] = lut[i * NHH + hh];
  __syncthreads();
  int g = lane >> 4, n = lane & 15;
  int l0 = qt << 4;
  const float SC = 0.044194173824159216f;  // 1/sqrt(512)
  const float* qrow = q + (size_t)(b * LL + l0 + n) * DMOD + hh * HDD;
  v16h qv = load_b16_f32(qrow, g, SC);      // B-operand == per-row q fragment
  float mrow = -INFINITY, lrow = 0.0f;
  v8f o0 = {}, o1 = {};
  int lq = l0 + n;
  const float* kbase = k + (size_t)b * LL * DMOD + hh * HDD;
  const float* vbase = v + (size_t)b * LL * DMOD + hh * HDD;
  for (int kt = 0; kt < LL / 32; ++kt) {
    int key0 = kt << 5;
    v16h kv0 = load_a16(kbase + (size_t)(key0 + n) * DMOD, g);
    v16h kv1 = load_a16(kbase + (size_t)(key0 + 16 + n) * DMOD, g);
    v8f zz = {};
    v8f st0 = wmma_f16(kv0, qv, zz);  // S^T tile, keys key0..key0+15
    v8f st1 = wmma_f16(kv1, qv, zz);  // keys key0+16..key0+31
    // add relative-position bias, track chunk max
    float cmax = -INFINITY;
#pragma unroll
    for (int r = 0; r < 8; ++r) {
      int key = key0 + 8 * g + r;
      int d1 = key - lq; d1 = d1 < -PMAXX ? -PMAXX : (d1 > PMAXX ? PMAXX : d1);
      st0[r] += sl[d1 + PMAXX];
      cmax = fmaxf(cmax, st0[r]);
      int d2 = key + 16 - lq; d2 = d2 < -PMAXX ? -PMAXX : (d2 > PMAXX ? PMAXX : d2);
      st1[r] += sl[d2 + PMAXX];
      cmax = fmaxf(cmax, st1[r]);
    }
    cmax = fmaxf(cmax, __shfl_xor(cmax, 16));
    float mnew = fmaxf(mrow, cmax);
    float alpha = __expf(mrow - mnew);
    float csum = 0.0f;
    v16h pa;  // S^T D-layout == P A-layout after exp (the transpose trick)
#pragma unroll
    for (int r = 0; r < 8; ++r) {
      float p0 = __expf(st0[r] - mnew);
      float p1 = __expf(st1[r] - mnew);
      csum += p0 + p1;
      pa[r] = (_Float16)p0;
      pa[8 + r] = (_Float16)p1;
    }
    csum += __shfl_xor(csum, 16);
    lrow = lrow * alpha + csum;
    mrow = mnew;
#pragma unroll
    for (int r = 0; r < 8; ++r) {  // rescale accumulators (row q = 8g+r)
      float s = __shfl(alpha, 8 * g + r);
      o0[r] *= s;
      o1[r] *= s;
    }
    // V fragments (B-operand): lane col n -> V[key0+8g+j+8*(j/8)][n] / [16+n]
    v16h vb0, vb1;
#pragma unroll
    for (int j = 0; j < 16; ++j) {
      const float* vp = vbase + (size_t)(key0 + 8 * g + j + (j >= 8 ? 8 : 0)) * DMOD;
      vb0[j] = (_Float16)vp[n];
      vb1[j] = (_Float16)vp[16 + n];
    }
    o0 = wmma_f16(pa, vb0, o0);
    o1 = wmma_f16(pa, vb1, o1);
  }
  float linv = 1.0f / lrow;
#pragma unroll
  for (int r = 0; r < 8; ++r) {
    float s = __shfl(linv, 8 * g + r);
    float* op = o + (size_t)(b * LL + l0 + 8 * g + r) * DMOD + hh * HDD;
    op[n] = o0[r] * s;
    op[16 + n] = o1[r] * s;
  }
}

// ---------------- LayerNorm over E=256 (one block per row) ----------------
__global__ void __launch_bounds__(256)
ln_kernel(const float* __restrict__ x, const float* __restrict__ gm,
          const float* __restrict__ bt, float* __restrict__ y) {
  int row = blockIdx.x, t = threadIdx.x;
  float v = x[(size_t)row * EE + t];
  float s = v, q = v * v;
#pragma unroll
  for (int o = 16; o >= 1; o >>= 1) {
    s += __shfl_xor(s, o);
    q += __shfl_xor(q, o);
  }
  __shared__ float ls[8], lsq[8];
  int w = t >> 5;
  if ((t & 31) == 0) { ls[w] = s; lsq[w] = q; }
  __syncthreads();
  float S = 0.0f, Q = 0.0f;
#pragma unroll
  for (int i = 0; i < 8; ++i) { S += ls[i]; Q += lsq[i]; }
  float mean = S * (1.0f / EE);
  float var = Q * (1.0f / EE) - mean * mean;
  y[(size_t)row * EE + t] = (v - mean) * rsqrtf(var + 1e-5f) * gm[t] + bt[t];
}

// ---------------- embed stage 1: relu(z^T @ ex_w1 + b1) ----------------
__global__ void __launch_bounds__(256)
embed1_kernel(const float* __restrict__ z, const float* __restrict__ w1,
              const float* __restrict__ b1, float* __restrict__ out) {
  int row = blockIdx.x;          // b*L + l
  int t = threadIdx.x;           // output channel
  int b = row >> 9, l = row & (LL - 1);
  __shared__ float zr[NZ];
  if (t < NZ) zr[t] = z[((size_t)b * NZ + t) * LL + l];
  __syncthreads();
  float a = b1[t];
#pragma unroll
  for (int nzi = 0; nzi < NZ; ++nzi) a += zr[nzi] * w1[nzi * EE + t];
  out[(size_t)row * EE + t] = fmaxf(a, 0.0f);
}

// ---------------- pair-bias LUT: posTable(49x64) @ W2d + b2d ----------------
__global__ void blut_kernel(const float* __restrict__ pos,
                            const float* __restrict__ W2d,
                            const float* __restrict__ b2d,
                            float* __restrict__ lut) {
  int l = blockIdx.x, bin = blockIdx.y, hh = threadIdx.x;
  float s = b2d[l * NHH + hh];
  for (int p = 0; p < PDIMM; ++p)
    s += pos[bin * PDIMM + p] * W2d[((size_t)l * PDIMM + p) * NHH + hh];
  lut[((size_t)l * NBIN + bin) * NHH + hh] = s;
}

// ---------------- aa embedding via argmax over 20 channels ----------------
__global__ void aa_kernel(const float* __restrict__ x,
                          const float* __restrict__ aa_table,
                          float* __restrict__ eaa) {
  int row = blockIdx.x * blockDim.x + threadIdx.x;
  if (row >= ROWS) return;
  int b = row >> 9, l = row & (LL - 1);
  const float* xp = x + (size_t)b * 20 * LL + l;
  int best = 0;
  float bv = xp[0];
  for (int c = 1; c < 20; ++c) {
    float vv = xp[(size_t)c * LL];
    if (vv > bv) { bv = vv; best = c; }
  }
  for (int j = 0; j < E1DD; ++j)
    eaa[(size_t)row * E1DD + j] = aa_table[best * E1DD + j];
}

// ---------------- um = [h | e_aa] ----------------
__global__ void concat_kernel(const float* __restrict__ h,
                              const float* __restrict__ eaa,
                              float* __restrict__ um) {
  size_t idx = (size_t)blockIdx.x * blockDim.x + threadIdx.x;
  if (idx >= (size_t)ROWS * (EE + E1DD)) return;
  size_t row = idx / (EE + E1DD);
  int c = (int)(idx % (EE + E1DD));
  um[idx] = (c < EE) ? h[row * EE + c] : eaa[row * E1DD + (c - EE)];
}

// ---------------- transpose + f16 convert: (cnt,K,N) -> (cnt,N,K) ----------
__global__ void tconv_kernel(const float* __restrict__ src,
                             _Float16* __restrict__ dst, int K, int N,
                             long long tot) {
  long long idx = (long long)blockIdx.x * blockDim.x + threadIdx.x;
  if (idx >= tot) return;
  long long mk = (long long)K * N;
  int m = (int)(idx / mk);
  long long rem = idx % mk;
  int kd = (int)(rem / N), nd = (int)(rem % N);
  dst[(long long)m * mk + (long long)nd * K + kd] = (_Float16)src[idx];
}

// ---------------- final 256 -> 3 projection ----------------
__global__ void rproj_kernel(const float* __restrict__ t1,
                             const float* __restrict__ w2,
                             const float* __restrict__ b2,
                             float* __restrict__ r) {
  int idx = blockIdx.x * blockDim.x + threadIdx.x;
  if (idx >= ROWS * 3) return;
  int row = idx / 3, j = idx % 3;
  float s = b2[j];
  for (int kk = 0; kk < EE; ++kk) s += t1[(size_t)row * EE + kk] * w2[kk * 3 + j];
  r[idx] = s;
}

// ---------------- pairwise distance map ----------------
__global__ void __launch_bounds__(128)
dist_kernel(const float* __restrict__ r, float* __restrict__ out) {
  int bi = blockIdx.x;  // b*L + i
  int b = bi >> 9, i = bi & (LL - 1);
  float x0 = r[(size_t)bi * 3], y0 = r[(size_t)bi * 3 + 1], z0 = r[(size_t)bi * 3 + 2];
  for (int j = threadIdx.x; j < LL; j += blockDim.x) {
    const float* rj = r + ((size_t)b * LL + j) * 3;
    float dx = x0 - rj[0], dy = y0 - rj[1], dz = z0 - rj[2];
    out[((size_t)b * LL + i) * LL + j] = sqrtf(dx * dx + dy * dy + dz * dz + 1e-12f);
  }
}

// =====================================================================
extern "C" void kernel_launch(void* const* d_in, const int* in_sizes, int n_in,
                              void* d_out, int out_size, void* d_ws,
                              size_t ws_size, hipStream_t stream) {
  (void)in_sizes; (void)n_in; (void)out_size; (void)ws_size;
  const float* z        = (const float*)d_in[0];
  const float* x        = (const float*)d_in[1];
  const float* pos      = (const float*)d_in[2];
  const float* aa_table = (const float*)d_in[3];
  const float* ex_w1    = (const float*)d_in[4];
  const float* ex_b1    = (const float*)d_in[5];
  const float* ex_w2    = (const float*)d_in[6];
  const float* ex_b2    = (const float*)d_in[7];
  const float* Wq       = (const float*)d_in[8];
  const float* Wk       = (const float*)d_in[9];
  const float* Wv       = (const float*)d_in[10];
  const float* Wo       = (const float*)d_in[11];
  const float* bo       = (const float*)d_in[12];
  const float* W2d      = (const float*)d_in[13];
  const float* b2d      = (const float*)d_in[14];
  const float* W1       = (const float*)d_in[15];
  const float* b1       = (const float*)d_in[16];
  const float* W2       = (const float*)d_in[17];
  const float* b2       = (const float*)d_in[18];
  const float* g1       = (const float*)d_in[19];
  const float* be1      = (const float*)d_in[20];
  const float* g2       = (const float*)d_in[21];
  const float* be2      = (const float*)d_in[22];
  const float* m3_w1    = (const float*)d_in[23];
  const float* m3_b1    = (const float*)d_in[24];
  const float* m3_w2    = (const float*)d_in[25];
  const float* m3_b2    = (const float*)d_in[26];

  // ---- workspace carve-up ----
  char* base = (char*)d_ws;
  size_t off = 0;
  auto falloc = [&](size_t nf) {
    float* p = (float*)(base + off);
    off += ((nf * 4 + 255) / 256) * 256;
    return p;
  };
  float* h   = falloc((size_t)ROWS * EE);
  float* h2  = falloc((size_t)ROWS * EE);
  float* qb  = falloc((size_t)ROWS * DMOD);
  float* kb  = falloc((size_t)ROWS * DMOD);
  float* vb  = falloc((size_t)ROWS * DMOD);
  float* ob  = falloc((size_t)ROWS * DMOD);
  float* um  = falloc((size_t)ROWS * (EE + E1DD));
  float* t1  = falloc((size_t)ROWS * FFF);
  float* eaa = falloc((size_t)ROWS * E1DD);
  float* rr  = falloc((size_t)ROWS * 3);
  float* lut = falloc((size_t)NLL * NBIN * NHH);
  auto halloc = [&](size_t nh) {
    _Float16* p = (_Float16*)(base + off);
    off += ((nh * 2 + 255) / 256) * 256;
    return p;
  };
  _Float16* WqT   = halloc((size_t)NLL * DMOD * EE);
  _Float16* WkT   = halloc((size_t)NLL * DMOD * EE);
  _Float16* WvT   = halloc((size_t)NLL * DMOD * EE);
  _Float16* WoT   = halloc((size_t)NLL * EE * DMOD);
  _Float16* W1T   = halloc((size_t)NLL * FFF * (EE + E1DD));
  _Float16* W2T   = halloc((size_t)NLL * EE * FFF);
  _Float16* exw2T = halloc((size_t)EE * EE);
  _Float16* m3w1T = halloc((size_t)EE * EE);

  auto tc = [&](const float* s, _Float16* d, int K, int N, int cnt) {
    long long tot = (long long)cnt * K * N;
    tconv_kernel<<<(unsigned)((tot + 255) / 256), 256, 0, stream>>>(s, d, K, N, tot);
  };
  auto gemm = [&](const float* A, int lda, const _Float16* Bt, int ldb, float* C,
                  int ldc, const float* bias, const float* resid, int M, int N,
                  int K, int relu) {
    gemm_wmma_kernel<<<dim3(N / 64, M / 32), 32, 0, stream>>>(
        A, lda, Bt, ldb, C, ldc, bias, resid, K, relu);
  };

  // ---- one-time weight transforms (re-run every call; deterministic) ----
  tc(Wq, WqT, EE, DMOD, NLL);
  tc(Wk, WkT, EE, DMOD, NLL);
  tc(Wv, WvT, EE, DMOD, NLL);
  tc(Wo, WoT, DMOD, EE, NLL);
  tc(W1, W1T, EE + E1DD, FFF, NLL);
  tc(W2, W2T, FFF, EE, NLL);
  tc(ex_w2, exw2T, EE, EE, 1);
  tc(m3_w1, m3w1T, EE, EE, 1);

  blut_kernel<<<dim3(NLL, NBIN), NHH, 0, stream>>>(pos, W2d, b2d, lut);
  aa_kernel<<<ROWS / 256, 256, 0, stream>>>(x, aa_table, eaa);

  // ---- embed ----
  embed1_kernel<<<ROWS, EE, 0, stream>>>(z, ex_w1, ex_b1, t1);
  gemm(t1, EE, exw2T, EE, h, EE, ex_b2, nullptr, ROWS, EE, EE, 0);

  // ---- transformer layers ----
  for (int i = 0; i < NLL; ++i) {
    gemm(h, EE, WqT + (size_t)i * DMOD * EE, EE, qb, DMOD, nullptr, nullptr,
         ROWS, DMOD, EE, 0);
    gemm(h, EE, WkT + (size_t)i * DMOD * EE, EE, kb, DMOD, nullptr, nullptr,
         ROWS, DMOD, EE, 0);
    gemm(h, EE, WvT + (size_t)i * DMOD * EE, EE, vb, DMOD, nullptr, nullptr,
         ROWS, DMOD, EE, 0);
    attn_kernel<<<dim3(LL / 16, NHH, BB), 32, 0, stream>>>(
        qb, kb, vb, lut + (size_t)i * NBIN * NHH, ob);
    gemm(ob, DMOD, WoT + (size_t)i * EE * DMOD, DMOD, h2, EE, bo + i * EE, h,
         ROWS, EE, DMOD, 0);
    ln_kernel<<<ROWS, EE, 0, stream>>>(h2, g1 + i * EE, be1 + i * EE, h);
    concat_kernel<<<(ROWS * (EE + E1DD) + 255) / 256, 256, 0, stream>>>(h, eaa, um);
    gemm(um, EE + E1DD, W1T + (size_t)i * FFF * (EE + E1DD), EE + E1DD, t1, FFF,
         b1 + i * FFF, nullptr, ROWS, FFF, EE + E1DD, 1);
    gemm(t1, FFF, W2T + (size_t)i * EE * FFF, FFF, h2, EE, b2 + i * EE, h,
         ROWS, EE, FFF, 0);
    ln_kernel<<<ROWS, EE, 0, stream>>>(h2, g2 + i * EE, be2 + i * EE, h);
  }

  // ---- head: 3-D coords + distance map ----
  gemm(h, EE, m3w1T, EE, t1, EE, m3_b1, nullptr, ROWS, EE, EE, 1);
  rproj_kernel<<<(ROWS * 3 + 255) / 256, 256, 0, stream>>>(t1, m3_w2, m3_b2, rr);
  dist_kernel<<<ROWS, 128, 0, stream>>>(rr, (float*)d_out);
}
